// DynamicModalFusion_75703093559465
// MI455X (gfx1250) — matrix operand order repaired and checked by reading, not compiled, not run on hardware
//
#include <hip/hip_runtime.h>
#include <stdint.h>

#define Bsz 4
#define Tn 512
#define Hn 2048
#define NH 16
#define HD 128
#define Cn (12*Hn)
#define ROWS (Bsz*Tn)
#define EPSf 1e-5f
#define SOFT_SCALE 0.08838834764831845f  /* 1/sqrt(128) */

typedef __attribute__((ext_vector_type(16))) __bf16 v16bf;
typedef __attribute__((ext_vector_type(8)))  float  v8f;

union FragB { v16bf v; unsigned int d[8]; unsigned short s[16]; };

__device__ __forceinline__ unsigned short f2bf(float f) {
    unsigned int u = __float_as_uint(f);
    u += 0x7fffu + ((u >> 16) & 1u);          // round-to-nearest-even
    return (unsigned short)(u >> 16);
}

// K index pattern for 16-bit A fragments (ISA 7.12.2): dword j holds K pair.
__device__ __forceinline__ int kpat(int j, int hi) {
    return ((j & 4) << 2) + 2 * (j & 3) + hi * 8;
}

__device__ __forceinline__ float xormax16(float v) {
    for (int m = 1; m < 16; m <<= 1) v = fmaxf(v, __shfl_xor(v, m, 32));
    return v;
}
__device__ __forceinline__ float xorsum16(float v) {
    for (int m = 1; m < 16; m <<= 1) v += __shfl_xor(v, m, 32);
    return v;
}

// --- CDNA5 async copy: global -> LDS, 16 bytes per lane (ASYNCcnt path) ----
__device__ __forceinline__ void async_copy16(unsigned int lds_off, const void* gptr) {
    unsigned long long ga = (unsigned long long)gptr;
    asm volatile("global_load_async_to_lds_b128 %0, %1, off"
                 :: "v"(lds_off), "v"(ga) : "memory");
}
__device__ __forceinline__ void async_wait0() {
    asm volatile("s_wait_asynccnt 0" ::: "memory");
}
// LDS byte offset of a shared-memory pointer (aperture: low 32 bits of the address)
__device__ __forceinline__ unsigned int lds_off_of(const void* p) {
    return (unsigned int)(unsigned long long)p;
}

// ---------------------------------------------------------------------------
// K0: cast one [H,H] weight matrix fp32 -> bf16 (b128 in / b128 out)
// ---------------------------------------------------------------------------
__global__ __launch_bounds__(256)
void castw_kernel(const float* __restrict__ W, unsigned short* __restrict__ Wb) {
    size_t i = ((size_t)blockIdx.x * 256 + threadIdx.x) * 8;
    const float4* p = (const float4*)(W + i);
    float4 a = p[0], b = p[1];
    uint4 r;
    r.x = (unsigned)f2bf(a.x) | ((unsigned)f2bf(a.y) << 16);
    r.y = (unsigned)f2bf(a.z) | ((unsigned)f2bf(a.w) << 16);
    r.z = (unsigned)f2bf(b.x) | ((unsigned)f2bf(b.y) << 16);
    r.w = (unsigned)f2bf(b.z) | ((unsigned)f2bf(b.w) << 16);
    *(uint4*)(Wb + i) = r;
}

// ---------------------------------------------------------------------------
// K1: per-row mean/rstd for all 6 modality tensors (LN stats, param-free)
// ---------------------------------------------------------------------------
__global__ __launch_bounds__(256)
void rowstats_kernel(const float* t0, const float* t1, const float* t2,
                     const float* t3, const float* t4, const float* t5,
                     float* __restrict__ stats) {
    int row = blockIdx.x;            // 0..ROWS-1
    int ten = blockIdx.y;            // 0..5
    const float* X = ten == 0 ? t0 : ten == 1 ? t1 : ten == 2 ? t2 :
                     ten == 3 ? t3 : ten == 4 ? t4 : t5;
    const float* xr = X + (size_t)row * Hn;
    float s = 0.f, ss = 0.f;
    for (int c = threadIdx.x; c < Hn; c += 256) { float v = xr[c]; s += v; ss += v * v; }
    __shared__ float rs[256], rss[256];
    rs[threadIdx.x] = s; rss[threadIdx.x] = ss; __syncthreads();
    for (int o = 128; o > 0; o >>= 1) {
        if (threadIdx.x < o) { rs[threadIdx.x] += rs[threadIdx.x + o];
                               rss[threadIdx.x] += rss[threadIdx.x + o]; }
        __syncthreads();
    }
    if (threadIdx.x == 0) {
        float mean = rs[0] * (1.0f / Hn);
        float var  = rss[0] * (1.0f / Hn) - mean * mean;
        stats[((size_t)ten * ROWS + row) * 2 + 0] = mean;
        stats[((size_t)ten * ROWS + row) * 2 + 1] = rsqrtf(var + EPSf);
    }
}

// ---------------------------------------------------------------------------
// K2: Y(bf16) = LayerNorm(X) @ W + bias, via v_wmma_f32_16x16x32_bf16
//     tile 128x128x32, 8 waves, each wave 2x4 sub-tiles.
//     A tile: VALU path (fused LN + cvt).  B tile: async global->LDS copy.
// ---------------------------------------------------------------------------
#define BM 128
#define BN 128
#define BK 32
#define ASTR 40
#define BSTR 136   /* 272 B row stride: 16B-aligned segments */

__global__ __launch_bounds__(256)
void gemm_ln_bf16(const float* __restrict__ X, const float* __restrict__ stats,
                  const float* __restrict__ lng, const float* __restrict__ lnb,
                  const unsigned short* __restrict__ Wbf,
                  const float* __restrict__ bias,
                  unsigned short* __restrict__ Y) {
    __shared__ __align__(16) unsigned short As[BM * ASTR];
    __shared__ __align__(16) unsigned short Bs[BK * BSTR];
    int tid = threadIdx.x;
    int lane = tid & 31, wid = tid >> 5;
    int wr = wid >> 1, wc = wid & 1;
    int hi = lane >> 4, lr = lane & 15;
    int m0 = blockIdx.y * BM, n0 = blockIdx.x * BN;
    unsigned int bs_base = lds_off_of(&Bs[0]);

    v8f acc[2][4];
    for (int m = 0; m < 2; m++)
        for (int n = 0; n < 4; n++)
            for (int r = 0; r < 8; r++) acc[m][n][r] = 0.f;

    for (int k0 = 0; k0 < Hn; k0 += BK) {
        // B tile (32x128 bf16): async copy, 2 x 16B chunks per thread
        for (int i = 0; i < 2; i++) {
            int c = tid + i * 256;            // 0..511
            int kk = c >> 4, seg = c & 15;    // row 0..31, 16B segment 0..15
            async_copy16(bs_base + (unsigned)(kk * BSTR + seg * 8) * 2,
                         &Wbf[(size_t)(k0 + kk) * Hn + n0 + seg * 8]);
        }
        // A tile (128x32) with fused LayerNorm + bf16 convert
        for (int i = 0; i < 16; i++) {
            int idx = tid + i * 256;
            int r = idx >> 5, k = idx & 31;
            int gr = m0 + r, gk = k0 + k;
            float x = X[(size_t)gr * Hn + gk];
            float a = (x - stats[gr * 2]) * stats[gr * 2 + 1] * lng[gk] + lnb[gk];
            As[r * ASTR + k] = f2bf(a);
        }
        if (k0 + BK < Hn)   // -> global_prefetch_b8
            __builtin_prefetch(&X[(size_t)(m0 + (tid >> 1)) * Hn + k0 + BK], 0, 1);
        async_wait0();
        __syncthreads();

        FragB afr[2], bfr[4];
        for (int m = 0; m < 2; m++) {
            int arow = wr * 32 + m * 16 + lr;
            for (int j = 0; j < 8; j++)
                afr[m].d[j] = *(const unsigned int*)&As[arow * ASTR + kpat(j, hi)];
        }
        for (int n = 0; n < 4; n++) {
            for (int j = 0; j < 8; j++)
                bfr[n].d[j] = (unsigned int)Bs[kpat(j, hi) * BSTR + wc * 64 + n * 16 + lr] |
                              ((unsigned int)Bs[(kpat(j, hi) + 1) * BSTR + wc * 64 + n * 16 + lr] << 16);
        }
        for (int m = 0; m < 2; m++)
            for (int n = 0; n < 4; n++)
                acc[m][n] = __builtin_amdgcn_wmma_f32_16x16x32_bf16(
                    false, afr[m].v, false, bfr[n].v, (short)0, acc[m][n], false, false);
        __syncthreads();
    }

    // epilogue: + bias, bf16 store (C/D layout: vgpr r -> row r + 8*hi, lane&15 = col)
    for (int m = 0; m < 2; m++)
        for (int n = 0; n < 4; n++) {
            int colg = n0 + wc * 64 + n * 16 + lr;
            float bcol = bias[colg];
            int rowbase = m0 + wr * 32 + m * 16 + hi * 8;
            for (int r = 0; r < 8; r++)
                Y[(size_t)(rowbase + r) * Hn + colg] = f2bf(acc[m][n][r] + bcol);
        }
}

// ---------------------------------------------------------------------------
// K3: per-(b,h): two-pass online softmax of Q K^T / sqrt(d), emit column means
//     s[b,h,k] = mean_t softmax(scores)[t,k].  K tile async-staged in LDS.
// ---------------------------------------------------------------------------
#define KSTR 136   /* 272 B row stride: 16B-aligned segments */

__device__ __forceinline__ void score_tile(const unsigned short* Ks, const FragB* aq,
                                           int kt, int lane, float* sv) {
    v8f c;
    for (int r = 0; r < 8; r++) c[r] = 0.f;
    for (int dc = 0; dc < 4; dc++) {
        FragB bk;
        for (int j = 0; j < 8; j++) {
            unsigned int e0 = Ks[(kt * 16 + (lane & 15)) * KSTR + dc * 32 + kpat(j, lane >> 4)];
            unsigned int e1 = Ks[(kt * 16 + (lane & 15)) * KSTR + dc * 32 + kpat(j, lane >> 4) + 1];
            bk.d[j] = e0 | (e1 << 16);
        }
        c = __builtin_amdgcn_wmma_f32_16x16x32_bf16(
            false, aq[dc].v, false, bk.v, (short)0, c, false, false);
    }
    for (int r = 0; r < 8; r++) sv[r] = c[r] * SOFT_SCALE;
}

__global__ __launch_bounds__(256)
void attn_stats(const unsigned short* __restrict__ Qbf,
                const unsigned short* __restrict__ Kbf,
                float* __restrict__ sOut) {
    extern __shared__ char smem[];
    unsigned short* Ks = (unsigned short*)smem;                       // 512*KSTR
    float* sacc = (float*)(smem + (size_t)Tn * KSTR * sizeof(unsigned short));
    int tid = threadIdx.x, lane = tid & 31, wid = tid >> 5;
    int hi = lane >> 4, lr = lane & 15;
    int b = blockIdx.x / NH, h = blockIdx.x % NH;
    size_t base = ((size_t)b * Tn) * Hn + (size_t)h * HD;
    unsigned int ks_base = lds_off_of(Ks);

    // async-stage K tile: 512 rows x 256 B, 16B chunks (32 per thread)
    for (int i = 0; i < 32; i++) {
        int c = tid + i * 256;           // 0..8191
        int t = c >> 4, seg = c & 15;
        async_copy16(ks_base + (unsigned)(t * KSTR + seg * 8) * 2,
                     &Kbf[base + (size_t)t * Hn + seg * 8]);
    }
    for (int k = tid; k < Tn; k += 256) sacc[k] = 0.f;
    async_wait0();
    __syncthreads();

    for (int qt = 0; qt < 4; qt++) {
        int qbase = (wid * 4 + qt) * 16;
        FragB aq[4];
        for (int dc = 0; dc < 4; dc++)
            for (int j = 0; j < 8; j++)
                aq[dc].d[j] = *(const unsigned int*)
                    &Qbf[base + (size_t)(qbase + lr) * Hn + dc * 32 + kpat(j, hi)];

        float rmax[8], rsum[8];
        for (int r = 0; r < 8; r++) { rmax[r] = -3.0e38f; rsum[r] = 0.f; }

        // pass 1: online row max / row sum
        for (int kt = 0; kt < 32; kt++) {
            float sv[8];
            score_tile(Ks, aq, kt, lane, sv);
            for (int r = 0; r < 8; r++) {
                float tm = xormax16(sv[r]);
                float nm = fmaxf(rmax[r], tm);
                float e  = __expf(sv[r] - nm);
                float ts = xorsum16(e);
                rsum[r] = rsum[r] * __expf(rmax[r] - nm) + ts;
                rmax[r] = nm;
            }
        }
        float rinv[8];
        for (int r = 0; r < 8; r++) rinv[r] = 1.f / rsum[r];

        // pass 2: recompute, normalize, accumulate column means
        for (int kt = 0; kt < 32; kt++) {
            float sv[8];
            score_tile(Ks, aq, kt, lane, sv);
            float part = 0.f;
            for (int r = 0; r < 8; r++) part += __expf(sv[r] - rmax[r]) * rinv[r];
            atomicAdd(&sacc[kt * 16 + lr], part * (1.0f / Tn));
        }
    }
    __syncthreads();
    for (int k = tid; k < Tn; k += 256)
        sOut[(size_t)blockIdx.x * Tn + k] = sacc[k];
}

// ---------------------------------------------------------------------------
// K4: u[b,h,:] = s[b,h,:] @ kv[b]      (T-contraction GEMV)
// ---------------------------------------------------------------------------
__global__ __launch_bounds__(256)
void s_kv_kernel(const float* __restrict__ sIn, const float* __restrict__ kv,
                 float* __restrict__ u) {
    int bh = blockIdx.x;
    int b = bh / NH;
    int c = blockIdx.y * 256 + threadIdx.x;
    __shared__ float sl[Tn];
    for (int k = threadIdx.x; k < Tn; k += 256) sl[k] = sIn[(size_t)bh * Tn + k];
    __syncthreads();
    float acc = 0.f;
    for (int k = 0; k < Tn; k++) acc += sl[k] * kv[((size_t)b * Tn + k) * Hn + c];
    u[(size_t)bh * Hn + c] = acc;
}

// ---------------------------------------------------------------------------
// K5: y0[b,col] = u[b,col>>7,:] @ Wv[:,col] + bv[col]   (mean attn pre-Wo)
// ---------------------------------------------------------------------------
__global__ __launch_bounds__(256)
void vproj_kernel(const float* __restrict__ u, const float* __restrict__ Wv,
                  const float* __restrict__ bv, float* __restrict__ y0) {
    int col = blockIdx.x * 256 + threadIdx.x;
    int h = col >> 7;
    for (int b = 0; b < Bsz; b++) {
        const float* ub = u + ((size_t)b * NH + h) * Hn;
        float acc = 0.f;
        for (int c2 = 0; c2 < Hn; c2++) acc += ub[c2] * Wv[(size_t)c2 * Hn + col];
        y0[(size_t)b * Hn + col] = acc + bv[col];
    }
}

// ---------------------------------------------------------------------------
// K6: mvec[b, 6H + mod*H + col] = 0.2 * (y0[b,:] @ Wo[:,col] + bo[col])
// ---------------------------------------------------------------------------
__global__ __launch_bounds__(256)
void oproj_kernel(const float* __restrict__ y0, const float* __restrict__ Wo,
                  const float* __restrict__ bo, float* __restrict__ mvec, int mod) {
    int col = blockIdx.x * 256 + threadIdx.x;
    for (int b = 0; b < Bsz; b++) {
        const float* yb = y0 + (size_t)b * Hn;
        float acc = 0.f;
        for (int a = 0; a < Hn; a++) acc += yb[a] * Wo[(size_t)a * Hn + col];
        mvec[(size_t)b * Cn + 6 * Hn + (size_t)mod * Hn + col] = 0.2f * (acc + bo[col]);
    }
}

// ---------------------------------------------------------------------------
// K7: mvec[b, i*H + col] = softmax(mw)[i] * mean_t(X_i[b,:,col])
// ---------------------------------------------------------------------------
__global__ __launch_bounds__(256)
void modmean_kernel(const float* t0, const float* t1, const float* t2,
                    const float* t3, const float* t4, const float* t5,
                    const float* __restrict__ mw, float* __restrict__ mvec) {
    int col = blockIdx.x * 256 + threadIdx.x;
    int i = blockIdx.y;
    const float* X = i == 0 ? t0 : i == 1 ? t1 : i == 2 ? t2 :
                     i == 3 ? t3 : i == 4 ? t4 : t5;
    float w[6]; float mx = -3.0e38f;
    for (int j = 0; j < 6; j++) { w[j] = mw[j]; mx = fmaxf(mx, w[j]); }
    float ssum = 0.f;
    for (int j = 0; j < 6; j++) { w[j] = __expf(w[j] - mx); ssum += w[j]; }
    float wi = w[i] / ssum;
    for (int b = 0; b < Bsz; b++) {
        float acc = 0.f;
        for (int t = 0; t < Tn; t++) acc += X[((size_t)b * Tn + t) * Hn + col];
        mvec[(size_t)b * Cn + (size_t)i * Hn + col] = wi * acc * (1.0f / Tn);
    }
}

// ---------------------------------------------------------------------------
// K8b: mean/rstd of mvec rows (final LayerNorm over 12H)
// ---------------------------------------------------------------------------
__global__ __launch_bounds__(256)
void mstats_kernel(const float* __restrict__ mvec, float* __restrict__ ms) {
    int b = blockIdx.x;
    const float* row = mvec + (size_t)b * Cn;
    float s = 0.f, ss = 0.f;
    for (int c = threadIdx.x; c < Cn; c += 256) { float v = row[c]; s += v; ss += v * v; }
    __shared__ float rs[256], rss[256];
    rs[threadIdx.x] = s; rss[threadIdx.x] = ss; __syncthreads();
    for (int o = 128; o > 0; o >>= 1) {
        if (threadIdx.x < o) { rs[threadIdx.x] += rs[threadIdx.x + o];
                               rss[threadIdx.x] += rss[threadIdx.x + o]; }
        __syncthreads();
    }
    if (threadIdx.x == 0) {
        float mean = rs[0] * (1.0f / Cn);
        float var  = rss[0] * (1.0f / Cn) - mean * mean;
        ms[b * 2] = mean; ms[b * 2 + 1] = rsqrtf(var + EPSf);
    }
}

// ---------------------------------------------------------------------------
// K8: hidden = silu(mvec @ W1 + b1)
// ---------------------------------------------------------------------------
__global__ __launch_bounds__(256)
void gate1_kernel(const float* __restrict__ mvec, const float* __restrict__ W1,
                  const float* __restrict__ b1, float* __restrict__ hid) {
    int j = blockIdx.x * 256 + threadIdx.x;
    float a0 = 0.f, a1 = 0.f, a2 = 0.f, a3 = 0.f;
    for (int c = 0; c < Cn; c++) {
        float wv = W1[(size_t)c * Hn + j];
        a0 += mvec[c] * wv;
        a1 += mvec[(size_t)Cn + c] * wv;
        a2 += mvec[2 * (size_t)Cn + c] * wv;
        a3 += mvec[3 * (size_t)Cn + c] * wv;
    }
    float bb = b1[j];
    float a[4] = { a0 + bb, a1 + bb, a2 + bb, a3 + bb };
    for (int b = 0; b < Bsz; b++) {
        float x = a[b];
        hid[(size_t)b * Hn + j] = x / (1.f + __expf(-x));
    }
}

// ---------------------------------------------------------------------------
// K9: out = layer_norm(mvec) * sigmoid(hidden @ W2 + b2)
// ---------------------------------------------------------------------------
__global__ __launch_bounds__(256)
void gate2_kernel(const float* __restrict__ mvec, const float* __restrict__ hid,
                  const float* __restrict__ W2, const float* __restrict__ b2,
                  const float* __restrict__ ng, const float* __restrict__ nb,
                  const float* __restrict__ ms, float* __restrict__ out) {
    int c = blockIdx.x * 256 + threadIdx.x;
    for (int b = 0; b < Bsz; b++) {
        const float* hb = hid + (size_t)b * Hn;
        float acc = 0.f;
        for (int j = 0; j < Hn; j++) acc += hb[j] * W2[(size_t)j * Cn + c];
        float gate = 1.f / (1.f + __expf(-(acc + b2[c])));
        float mval = mvec[(size_t)b * Cn + c];
        float ln = (mval - ms[b * 2]) * ms[b * 2 + 1] * ng[c] + nb[c];
        out[(size_t)b * Cn + c] = ln * gate;
    }
}

// ---------------------------------------------------------------------------
extern "C" void kernel_launch(void* const* d_in, const int* in_sizes, int n_in,
                              void* d_out, int out_size, void* d_ws, size_t ws_size,
                              hipStream_t stream) {
    (void)in_sizes; (void)n_in; (void)out_size; (void)ws_size;
    const float* tens[6];
    for (int i = 0; i < 6; i++) tens[i] = (const float*)d_in[i];
    const float* Wq   = (const float*)d_in[6];
    const float* Wk   = (const float*)d_in[7];
    const float* Wv   = (const float*)d_in[8];
    const float* Wo   = (const float*)d_in[9];
    const float* bq   = (const float*)d_in[10];
    const float* bk   = (const float*)d_in[11];
    const float* bv   = (const float*)d_in[12];
    const float* bo   = (const float*)d_in[13];
    const float* ln1g = (const float*)d_in[14];
    const float* ln1b = (const float*)d_in[15];
    const float* ln2g = (const float*)d_in[16];
    const float* ln2b = (const float*)d_in[17];
    const float* mw   = (const float*)d_in[18];
    const float* W1   = (const float*)d_in[19];
    const float* b1   = (const float*)d_in[20];
    const float* W2   = (const float*)d_in[21];
    const float* b2   = (const float*)d_in[22];
    const float* ng   = (const float*)d_in[23];
    const float* nb   = (const float*)d_in[24];

    char* wsp = (char*)d_ws;
    auto carve = [&](size_t bytes) -> char* {
        char* p = wsp; wsp += (bytes + 255) & ~(size_t)255; return p;
    };
    float*          stats = (float*)carve((size_t)6 * ROWS * 2 * 4);
    unsigned short* Qbf   = (unsigned short*)carve((size_t)ROWS * Hn * 2);
    unsigned short* Kbf   = (unsigned short*)carve((size_t)ROWS * Hn * 2);
    unsigned short* Wqbf  = (unsigned short*)carve((size_t)Hn * Hn * 2);
    unsigned short* Wkbf  = (unsigned short*)carve((size_t)Hn * Hn * 2);
    float*          sbuf  = (float*)carve((size_t)Bsz * NH * Tn * 4);
    float*          ubuf  = (float*)carve((size_t)Bsz * NH * Hn * 4);
    float*          y0    = (float*)carve((size_t)Bsz * Hn * 4);
    float*          mvec  = (float*)carve((size_t)Bsz * Cn * 4);
    float*          hid   = (float*)carve((size_t)Bsz * Hn * 4);
    float*          ms    = (float*)carve((size_t)Bsz * 2 * 4);

    rowstats_kernel<<<dim3(ROWS, 6), 256, 0, stream>>>(
        tens[0], tens[1], tens[2], tens[3], tens[4], tens[5], stats);

    static const int qidx[6] = {0, 0, 0, 1, 1, 2};
    static const int kidx[6] = {1, 2, 5, 2, 5, 5};
    const int castGrid = (Hn * Hn) / (256 * 8);
    size_t attn_smem = (size_t)Tn * KSTR * sizeof(unsigned short) + Tn * sizeof(float);

    for (int mod = 0; mod < 6; mod++) {
        const float* qX = tens[qidx[mod]];
        const float* kX = tens[kidx[mod]];
        castw_kernel<<<castGrid, 256, 0, stream>>>(Wq + (size_t)mod * Hn * Hn, Wqbf);
        castw_kernel<<<castGrid, 256, 0, stream>>>(Wk + (size_t)mod * Hn * Hn, Wkbf);
        gemm_ln_bf16<<<dim3(Hn / BN, ROWS / BM), 256, 0, stream>>>(
            qX, stats + (size_t)qidx[mod] * ROWS * 2,
            ln1g + (size_t)mod * Hn, ln1b + (size_t)mod * Hn,
            Wqbf, bq + (size_t)mod * Hn, Qbf);
        gemm_ln_bf16<<<dim3(Hn / BN, ROWS / BM), 256, 0, stream>>>(
            kX, stats + (size_t)kidx[mod] * ROWS * 2,
            ln2g + (size_t)mod * Hn, ln2b + (size_t)mod * Hn,
            Wkbf, bk + (size_t)mod * Hn, Kbf);
        attn_stats<<<Bsz * NH, 256, attn_smem, stream>>>(Qbf, Kbf, sbuf);
        s_kv_kernel<<<dim3(Bsz * NH, Hn / 256), 256, 0, stream>>>(sbuf, kX, ubuf);
        vproj_kernel<<<Hn / 256, 256, 0, stream>>>(
            ubuf, Wv + (size_t)mod * Hn * Hn, bv + (size_t)mod * Hn, y0);
        oproj_kernel<<<Hn / 256, 256, 0, stream>>>(
            y0, Wo + (size_t)mod * Hn * Hn, bo + (size_t)mod * Hn, mvec, mod);
    }

    modmean_kernel<<<dim3(Hn / 256, 6), 256, 0, stream>>>(
        tens[0], tens[1], tens[2], tens[3], tens[4], tens[5], mw, mvec);
    mstats_kernel<<<Bsz, 256, 0, stream>>>(mvec, ms);
    gate1_kernel<<<Hn / 256, 256, 0, stream>>>(mvec, W1, b1, hid);
    gate2_kernel<<<Cn / 256, 256, 0, stream>>>(mvec, hid, W2, b2, ng, nb, ms,
                                               (float*)d_out);
}
